// _RotaryAttentionWrapper_12051678232736
// MI455X (gfx1250) — matrix operand order repaired and checked by reading, not compiled, not run on hardware
//
#include <hip/hip_runtime.h>
#include <hip/hip_bf16.h>

// ---------------------------------------------------------------------------
// MI455X (gfx1250) fused rotary-attention block.
// All matmuls run on v_wmma_f32_16x16x32_bf16 (wave32 WMMA).
// ---------------------------------------------------------------------------

typedef __bf16 bf16;
typedef __attribute__((ext_vector_type(16))) __bf16 v16bf;
typedef __attribute__((ext_vector_type(8)))  __bf16 v8bf;
typedef __attribute__((ext_vector_type(8)))  float  v8f;

union Frag {
  v16bf v;
  v8bf  h[2];
};

__device__ __forceinline__ v8f wmma_bf16(const Frag& a, const Frag& b, v8f c) {
  // 8 args: (neg_a, A, neg_b, B, c_mod, C, reuse_a, reuse_b)
  return __builtin_amdgcn_wmma_f32_16x16x32_bf16(
      false, a.v, false, b.v, (short)0, c, false, false);
}

// ---------------------------------------------------------------------------
// Elementwise f32 -> bf16
// ---------------------------------------------------------------------------
__global__ void cvt_f32_bf16(const float* __restrict__ in, bf16* __restrict__ out,
                             long n) {
  long i = (long)blockIdx.x * blockDim.x + threadIdx.x;
  long stride = (long)gridDim.x * blockDim.x;
  for (; i < n; i += stride) out[i] = (bf16)in[i];
}

// ---------------------------------------------------------------------------
// NT GEMM: C[m,n] = sum_k A[m,k] * B[n,k].  A:(M,K) bf16 row-major,
// B:(N,K) bf16 row-major (weight layout), C:(M,N) f32.
// Block = 256 threads = 8 waves; block tile 128x128; wave tile 32x64.
// Fragment loads are plain contiguous 16B loads matching the ISA layouts:
//   A (16x32, 16-bit): lane m%16, chunks at k = 8*hi and 16+8*hi
//   B (32x16, 16-bit): lane n%16, chunks at k = 8*hi and 16+8*hi
// ---------------------------------------------------------------------------
__global__ __launch_bounds__(256)
void gemm_nt_bf16(const bf16* __restrict__ A, const bf16* __restrict__ B,
                  float* __restrict__ C, int M, int N, int K) {
  const int lane = threadIdx.x & 31;
  const int wave = threadIdx.x >> 5;
  const int wm = wave & 3;         // 0..3 -> 32-row slab
  const int wn = wave >> 2;        // 0..1 -> 64-col slab
  const int m0 = blockIdx.x * 128 + wm * 32;
  const int n0 = blockIdx.y * 128 + wn * 64;
  const int r  = lane & 15;
  const int hi = lane >> 4;

  v8f acc[2][4] = {};

  for (int kb = 0; kb < K; kb += 32) {
    Frag a[2], b[4];
#pragma unroll
    for (int i = 0; i < 2; ++i) {
      const bf16* p = A + (size_t)(m0 + 16 * i + r) * K + kb + 8 * hi;
      a[i].h[0] = *(const v8bf*)p;
      a[i].h[1] = *(const v8bf*)(p + 16);
    }
#pragma unroll
    for (int j = 0; j < 4; ++j) {
      const bf16* p = B + (size_t)(n0 + 16 * j + r) * K + kb + 8 * hi;
      b[j].h[0] = *(const v8bf*)p;
      b[j].h[1] = *(const v8bf*)(p + 16);
    }
#pragma unroll
    for (int i = 0; i < 2; ++i)
#pragma unroll
      for (int j = 0; j < 4; ++j)
        acc[i][j] = wmma_bf16(a[i], b[j], acc[i][j]);
  }

  // C/D layout: lanes 0-15 hold M=vgpr,N=lane; lanes 16-31 hold M=vgpr+8.
#pragma unroll
  for (int i = 0; i < 2; ++i)
#pragma unroll
    for (int j = 0; j < 4; ++j) {
      float* cp = C + (size_t)(m0 + 16 * i + 8 * hi) * N + n0 + 16 * j + r;
#pragma unroll
      for (int vr = 0; vr < 8; ++vr)
        cp[(size_t)vr * N] = acc[i][j][vr];
    }
}

// ---------------------------------------------------------------------------
// RoPE (interleaved-pair form) on f32 GEMM output, emitting bf16.
// cols = n_heads*128; cos/sin are (T, 64).
// ---------------------------------------------------------------------------
__global__ void rope_f32_bf16(const float* __restrict__ in,
                              const float* __restrict__ cs,
                              const float* __restrict__ sn,
                              bf16* __restrict__ out, int T, int cols) {
  long npairs = (long)T * (cols / 2);
  long i = (long)blockIdx.x * blockDim.x + threadIdx.x;
  long stride = (long)gridDim.x * blockDim.x;
  for (; i < npairs; i += stride) {
    int t = (int)(i / (cols / 2));
    int p = (int)(i % (cols / 2));
    int hp = p & 63;                     // pair index within head (HEAD_DIM/2)
    float c = cs[(size_t)t * 64 + hp];
    float s = sn[(size_t)t * 64 + hp];
    float x1 = in[(size_t)t * cols + 2 * p];
    float x2 = in[(size_t)t * cols + 2 * p + 1];
    out[(size_t)t * cols + 2 * p]     = (bf16)(x1 * c - x2 * s);
    out[(size_t)t * cols + 2 * p + 1] = (bf16)(x1 * s + x2 * c);
  }
}

// ---------------------------------------------------------------------------
// V: f32 (T, 1024) -> bf16 transposed (1024, T) so P@V B-fragments are
// contiguous along the key dimension.
// ---------------------------------------------------------------------------
__global__ void v_transpose_bf16(const float* __restrict__ in,
                                 bf16* __restrict__ out, int T, int C) {
  long n = (long)T * C;
  long i = (long)blockIdx.x * blockDim.x + threadIdx.x;
  long stride = (long)gridDim.x * blockDim.x;
  for (; i < n; i += stride) {
    int t = (int)(i / C);
    int c = (int)(i % C);
    out[(size_t)c * T + t] = (bf16)in[i];
  }
}

// ---------------------------------------------------------------------------
// Flash attention (causal, GQA rep=4). Grid: (T/64, 32 heads), block = 128
// threads (4 waves). Each wave owns 16 q rows; 32 keys per iteration.
// S = Q.K^T via 8 WMMAs; online softmax with cross-lane reductions over the
// 16-lane half that owns each row; P re-shaped C-layout -> A-fragment via
// per-wave LDS; P.V via 8 WMMAs into 16x128 f32 accumulators.
// ---------------------------------------------------------------------------
__global__ __launch_bounds__(128)
void flash_attn_bf16(const bf16* __restrict__ Q, const bf16* __restrict__ Kk,
                     const bf16* __restrict__ Vt, bf16* __restrict__ Y,
                     int T) {
  __shared__ bf16 lds_p[4][16][32];

  const int lane = threadIdx.x & 31;
  const int wave = threadIdx.x >> 5;
  const int h    = blockIdx.y;
  const int kvh  = h >> 2;                 // N_REP = 4
  const int q0   = blockIdx.x * 64 + wave * 16;
  const int r    = lane & 15;
  const int hi   = lane >> 4;
  const float scale = 0.08838834764831845f;   // 1/sqrt(128)
  const float log2e = 1.4426950408889634f;

  // Q fragments: 4 d-chunks of 32 (A-operand layout).
  Frag qf[4];
  {
    const bf16* qrow = Q + (size_t)(q0 + r) * 4096 + h * 128 + 8 * hi;
#pragma unroll
    for (int c = 0; c < 4; ++c) {
      qf[c].h[0] = *(const v8bf*)(qrow + 32 * c);
      qf[c].h[1] = *(const v8bf*)(qrow + 32 * c + 16);
    }
  }

  v8f o[8] = {};
  float mrow[8], lrow[8];
#pragma unroll
  for (int vr = 0; vr < 8; ++vr) { mrow[vr] = -3.0e38f; lrow[vr] = 0.0f; }

  const int kend = q0 + 16;                // keys 0 .. q0+15 are live
  for (int kb = 0; kb < kend; kb += 32) {
    // ---- S = Q.K^T (two 16x16 tiles over 32 keys) ----
    v8f s[2] = {};
#pragma unroll
    for (int j = 0; j < 2; ++j) {
      const bf16* krow = Kk + (size_t)(kb + 16 * j + r) * 1024 + kvh * 128 + 8 * hi;
#pragma unroll
      for (int c = 0; c < 4; ++c) {
        Frag kf;
        kf.h[0] = *(const v8bf*)(krow + 32 * c);
        kf.h[1] = *(const v8bf*)(krow + 32 * c + 16);
        s[j] = wmma_bf16(qf[c], kf, s[j]);
      }
    }

    // ---- scale + causal mask ----
#pragma unroll
    for (int j = 0; j < 2; ++j)
#pragma unroll
      for (int vr = 0; vr < 8; ++vr) {
        int row = q0 + 8 * hi + vr;
        int col = kb + 16 * j + r;
        float sv = s[j][vr] * scale;
        if (col > row) sv = -3.0e38f;
        s[j][vr] = sv;
      }

    // ---- online softmax: row max / exp / row sum ----
    float mnew[8], alpha[8];
#pragma unroll
    for (int vr = 0; vr < 8; ++vr) {
      float mx = fmaxf(s[0][vr], s[1][vr]);
#pragma unroll
      for (int stp = 1; stp < 16; stp <<= 1)
        mx = fmaxf(mx, __shfl_xor(mx, stp, 32));
      mnew[vr]  = fmaxf(mrow[vr], mx);
      alpha[vr] = __builtin_exp2f((mrow[vr] - mnew[vr]) * log2e);
      mrow[vr]  = mnew[vr];
    }
#pragma unroll
    for (int j = 0; j < 2; ++j)
#pragma unroll
      for (int vr = 0; vr < 8; ++vr)
        s[j][vr] = __builtin_exp2f((s[j][vr] - mnew[vr]) * log2e);
#pragma unroll
    for (int vr = 0; vr < 8; ++vr) {
      float ps = s[0][vr] + s[1][vr];
#pragma unroll
      for (int stp = 1; stp < 16; stp <<= 1)
        ps += __shfl_xor(ps, stp, 32);
      lrow[vr] = lrow[vr] * alpha[vr] + ps;
    }

    // ---- rescale accumulators ----
#pragma unroll
    for (int t8 = 0; t8 < 8; ++t8)
#pragma unroll
      for (int vr = 0; vr < 8; ++vr)
        o[t8][vr] *= alpha[vr];

    // ---- P: C-layout -> bf16 A-fragment through per-wave LDS ----
#pragma unroll
    for (int j = 0; j < 2; ++j)
#pragma unroll
      for (int vr = 0; vr < 8; ++vr)
        lds_p[wave][8 * hi + vr][16 * j + r] = (bf16)s[j][vr];
    __asm__ volatile("" ::: "memory");   // LDS ops are in-order within a wave
    Frag pf;
    pf.h[0] = *(const v8bf*)&lds_p[wave][r][8 * hi];
    pf.h[1] = *(const v8bf*)&lds_p[wave][r][16 + 8 * hi];

    // ---- O += P.V  (V transposed: (d, T)) ----
#pragma unroll
    for (int t8 = 0; t8 < 8; ++t8) {
      const bf16* vp = Vt + (size_t)(kvh * 128 + 16 * t8 + r) * T + kb + 8 * hi;
      Frag vf;
      vf.h[0] = *(const v8bf*)vp;
      vf.h[1] = *(const v8bf*)(vp + 16);
      o[t8] = wmma_bf16(pf, vf, o[t8]);
    }
  }

  // ---- epilogue: divide by row sums, emit bf16 ----
#pragma unroll
  for (int t8 = 0; t8 < 8; ++t8)
#pragma unroll
    for (int vr = 0; vr < 8; ++vr) {
      int row = q0 + 8 * hi + vr;
      Y[(size_t)row * 4096 + h * 128 + 16 * t8 + r] = (bf16)(o[t8][vr] / lrow[vr]);
    }
}

// ---------------------------------------------------------------------------
// Host-side orchestration
// ---------------------------------------------------------------------------
extern "C" void kernel_launch(void* const* d_in, const int* in_sizes, int n_in,
                              void* d_out, int out_size, void* d_ws, size_t ws_size,
                              hipStream_t stream) {
  (void)in_sizes; (void)n_in; (void)out_size; (void)ws_size;

  const int T = 2048, D = 4096, DKV = 1024;

  const float* x   = (const float*)d_in[0];
  const float* cs  = (const float*)d_in[1];
  const float* sn  = (const float*)d_in[2];
  const float* wq  = (const float*)d_in[3];
  const float* wk  = (const float*)d_in[4];
  const float* wv  = (const float*)d_in[5];
  const float* wo  = (const float*)d_in[6];
  float* out = (float*)d_out;

  // workspace carve-out
  char* w = (char*)d_ws;
  bf16* xb  = (bf16*)w;  w += (size_t)T * D * 2;
  bf16* wqb = (bf16*)w;  w += (size_t)D * D * 2;
  bf16* wkb = (bf16*)w;  w += (size_t)DKV * D * 2;
  bf16* wvb = (bf16*)w;  w += (size_t)DKV * D * 2;
  bf16* wob = (bf16*)w;  w += (size_t)D * D * 2;
  float* qf = (float*)w; w += (size_t)T * D * 4;
  float* kf = (float*)w; w += (size_t)T * DKV * 4;
  float* vf = (float*)w; w += (size_t)T * DKV * 4;
  bf16* qb  = (bf16*)w;  w += (size_t)T * D * 2;
  bf16* kb  = (bf16*)w;  w += (size_t)T * DKV * 2;
  bf16* vtb = (bf16*)w;  w += (size_t)DKV * T * 2;
  bf16* yb  = (bf16*)qf;  // qf is dead after RoPE; reuse for attention output

  const int CB = 256, CG = 2048;  // grid-stride elementwise launches

  // 1) f32 -> bf16 conversions
  cvt_f32_bf16<<<CG, CB, 0, stream>>>(x,  xb,  (long)T * D);
  cvt_f32_bf16<<<CG, CB, 0, stream>>>(wq, wqb, (long)D * D);
  cvt_f32_bf16<<<CG, CB, 0, stream>>>(wk, wkb, (long)DKV * D);
  cvt_f32_bf16<<<CG, CB, 0, stream>>>(wv, wvb, (long)DKV * D);
  cvt_f32_bf16<<<CG, CB, 0, stream>>>(wo, wob, (long)D * D);

  // 2) QKV projections (NT GEMM, WMMA bf16)
  gemm_nt_bf16<<<dim3(T / 128, D / 128),   256, 0, stream>>>(xb, wqb, qf, T, D,   D);
  gemm_nt_bf16<<<dim3(T / 128, DKV / 128), 256, 0, stream>>>(xb, wkb, kf, T, DKV, D);
  gemm_nt_bf16<<<dim3(T / 128, DKV / 128), 256, 0, stream>>>(xb, wvb, vf, T, DKV, D);

  // 3) RoPE on Q, K; transpose-convert V
  rope_f32_bf16<<<CG, CB, 0, stream>>>(qf, cs, sn, qb, T, D);
  rope_f32_bf16<<<CG, CB, 0, stream>>>(kf, cs, sn, kb, T, DKV);
  v_transpose_bf16<<<CG, CB, 0, stream>>>(vf, vtb, T, DKV);

  // 4) causal flash attention (WMMA bf16)
  flash_attn_bf16<<<dim3(T / 64, 32), 128, 0, stream>>>(qb, kb, vtb, yb, T);

  // 5) output projection -> f32 d_out
  gemm_nt_bf16<<<dim3(T / 128, D / 128), 256, 0, stream>>>(yb, wob, out, T, D, D);
}
